// DDSPVocoder_40338332844536
// MI455X (gfx1250) — compile-verified, matrix operand order checked by reading, not run on hardware
//
#include <hip/hip_runtime.h>
#include <math.h>

typedef __attribute__((ext_vector_type(16))) _Float16 v16h;
typedef __attribute__((ext_vector_type(8)))  _Float16 v8h;
typedef __attribute__((ext_vector_type(8)))  float    v8f;

#define PI_F 3.14159265358979323846f
#define SR_F 48000.0f
#define BATCH 2
#define TFRM 400
#define NSAMP 96000
#define TSTFT 376          // 1 + 96000/256
#define NFFT 1024
#define HOPN 256
#define PADN 512
#define NHEAD 1287         // 2*128 + 2 + 2*513 + 2 + 1
#define NHEADP 1312        // padded to multiple of 32
#define KFEAT 96           // 72 padded to multiple of 32

// ---------------------------------------------------------------- encoding
__global__ void k_encode(const float* __restrict__ f0, const int* __restrict__ vel,
                         float* __restrict__ feat) {
  int idx = blockIdx.x * blockDim.x + threadIdx.x;     // over B*T
  if (idx >= BATCH * TFRM) return;
  int b = idx / TFRM;
  float f  = f0[idx];
  float fs = fmaxf(f, 20.0f);
  float fn = (__logf(fs) - __logf(20.0f)) / (__logf(5000.0f) - __logf(20.0f));
  fn = fminf(fmaxf(fn, 0.0f), 1.0f);
  float gate = (f > 0.0f) ? 1.0f : 0.0f;
  float* o = feat + (size_t)idx * KFEAT;
  #pragma unroll 4
  for (int i = 1; i <= 32; ++i) {
    float ang = PI_F * (float)i * fn;
    o[(i - 1) * 2]     = __sinf(ang) * gate;
    o[(i - 1) * 2 + 1] = __cosf(ang) * gate;
  }
  float vn = fminf(fmaxf((float)vel[b] / 7.0f, 0.0f), 1.0f);
  #pragma unroll
  for (int i = 1; i <= 4; ++i) {
    float ang = PI_F * (float)i * vn;
    o[64 + (i - 1) * 2]     = __sinf(ang);
    o[64 + (i - 1) * 2 + 1] = __cosf(ang);
  }
  #pragma unroll
  for (int i = 72; i < KFEAT; ++i) o[i] = 0.0f;        // zero K-pad
}

// zero-padded copy of W_pre1: (72,128) -> (96,128)
__global__ void k_pad_w1(const float* __restrict__ W, float* __restrict__ Wp) {
  int i = blockIdx.x * blockDim.x + threadIdx.x;
  if (i >= KFEAT * 128) return;
  int k = i / 128, n = i % 128;
  Wp[i] = (k < 72) ? W[k * 128 + n] : 0.0f;
}

// ------------------------------------------------- WMMA GEMM  C = act(A@W+b)
// Requirements (guaranteed by padding): M%16==0, K%32==0, N%32==0.
// A:(M,K) f32 row-major.  W:(K,N) row-major, or (N,K) if wTrans.
// One wave computes a 16x32 output strip (two 16x16 WMMA tiles).
// Tiles staged through LDS as f16; fragments read as 16B LDS vectors.
__global__ void __launch_bounds__(32)
k_gemm_wmma(const float* __restrict__ A, const float* __restrict__ W,
            const float* __restrict__ bias, float* __restrict__ C,
            int M, int K, int N, int wTrans, int act) {
  __shared__ _Float16 lA[16 * 32];     // [m][k]
  __shared__ _Float16 lB[32 * 32];     // [n][k]
  int ntx  = N >> 5;                   // N/32 strips
  int tm   = blockIdx.x / ntx, tn = blockIdx.x % ntx;
  int row0 = tm * 16, col0 = tn * 32;
  int lane = threadIdx.x;
  int fm   = lane & 15;
  int koff = (lane >> 4) * 8;          // ISA 16-bit A/B fragment K sub-offset
  v8f acc0 = {}, acc1 = {};

  for (int k0 = 0; k0 < K; k0 += 32) {
    // ---- stage A tile (16x32): coalesced, no bounds checks
    #pragma unroll
    for (int i = 0; i < 16; ++i) {
      int idx = i * 32 + lane;                 // consecutive lanes -> consecutive k
      int r = idx >> 5, c = idx & 31;
      lA[idx] = (_Float16)A[(size_t)(row0 + r) * K + (k0 + c)];
    }
    // ---- stage B tile (32 cols x 32 k) as [n][k]
    if (wTrans) {            // W:(N,K) -> consecutive lanes read consecutive k
      #pragma unroll
      for (int i = 0; i < 32; ++i) {
        int idx = i * 32 + lane;
        int n = idx >> 5, c = idx & 31;
        lB[n * 32 + c] = (_Float16)W[(size_t)(col0 + n) * K + (k0 + c)];
      }
    } else {                 // W:(K,N) -> consecutive lanes read consecutive n
      #pragma unroll
      for (int i = 0; i < 32; ++i) {
        int idx = i * 32 + lane;
        int n = idx & 31, c = idx >> 5;
        lB[n * 32 + c] = (_Float16)W[(size_t)(k0 + c) * N + (col0 + n)];
      }
    }
    __syncthreads();

    // ---- assemble fragments: two ds_load_b128 each
    v8h a0 = *reinterpret_cast<const v8h*>(&lA[fm * 32 + koff]);
    v8h a1 = *reinterpret_cast<const v8h*>(&lA[fm * 32 + 16 + koff]);
    v8h b00 = *reinterpret_cast<const v8h*>(&lB[fm * 32 + koff]);
    v8h b01 = *reinterpret_cast<const v8h*>(&lB[fm * 32 + 16 + koff]);
    v8h b10 = *reinterpret_cast<const v8h*>(&lB[(16 + fm) * 32 + koff]);
    v8h b11 = *reinterpret_cast<const v8h*>(&lB[(16 + fm) * 32 + 16 + koff]);
    v16h av, bv0, bv1;
    #pragma unroll
    for (int e = 0; e < 8; ++e) {
      av[e] = a0[e];  av[e + 8] = a1[e];
      bv0[e] = b00[e]; bv0[e + 8] = b01[e];
      bv1[e] = b10[e]; bv1[e + 8] = b11[e];
    }
    acc0 = __builtin_amdgcn_wmma_f32_16x16x32_f16(false, av, false, bv0,
                                                  (short)0, acc0, false, false);
    acc1 = __builtin_amdgcn_wmma_f32_16x16x32_f16(false, av, false, bv1,
                                                  (short)0, acc1, false, false);
    __syncthreads();
  }

  int m  = row0 + (lane >= 16 ? 8 : 0);
  int n0 = col0 + (lane & 15);
  #pragma unroll
  for (int v = 0; v < 8; ++v) {
    float x0 = acc0[v] + bias[n0];
    float x1 = acc1[v] + bias[n0 + 16];
    if (act == 1) { x0 = fmaxf(x0, 0.0f); x1 = fmaxf(x1, 0.0f); }
    C[(size_t)(m + v) * N + n0]      = x0;
    C[(size_t)(m + v) * N + n0 + 16] = x1;
  }
}

// -------------------------------------------------------------------- GRU
__global__ void k_gru(const float* __restrict__ gi, const float* __restrict__ W_hh,
                      const float* __restrict__ b_hh, float* __restrict__ hs) {
  __shared__ float h[BATCH * 64];
  __shared__ float g[BATCH * 192];
  __shared__ float hnew[BATCH * 64];
  int tid = threadIdx.x;
  for (int i = tid; i < BATCH * 64; i += blockDim.x) h[i] = 0.0f;
  __syncthreads();
  for (int t = 0; t < TFRM; ++t) {
    if (tid < BATCH * 192) {
      int b = tid / 192, j = tid % 192;
      float s = b_hh[j];
      const float* wr = W_hh + (size_t)j * 64;
      const float* hb = h + b * 64;
      #pragma unroll 8
      for (int u = 0; u < 64; ++u) s += wr[u] * hb[u];
      g[tid] = s;
    }
    __syncthreads();
    if (tid < BATCH * 64) {
      int b = tid / 64, u = tid % 64;
      const float* gib = gi + ((size_t)b * TFRM + t) * 192;
      float r = 1.0f / (1.0f + __expf(-(gib[u]       + g[b * 192 + u])));
      float z = 1.0f / (1.0f + __expf(-(gib[64 + u]  + g[b * 192 + 64 + u])));
      float n = tanhf(gib[128 + u] + r * g[b * 192 + 128 + u]);
      float hv = (1.0f - z) * n + z * h[tid];
      hs[((size_t)b * TFRM + t) * 64 + u] = hv;
      hnew[tid] = hv;
    }
    __syncthreads();
    if (tid < BATCH * 64) h[tid] = hnew[tid];
    __syncthreads();
  }
}

// --------------------------------------------- pack all heads into one GEMM
// columns: [harmL(128) harmR(128) ampL ampR noiseL(513) noiseR(513) nampL nampR B pad(25)]
__global__ void k_pack_heads(const float* __restrict__ W_harm, const float* __restrict__ b_harm,
                             const float* __restrict__ W_amp,  const float* __restrict__ b_amp,
                             const float* __restrict__ W_noise,const float* __restrict__ b_noise,
                             const float* __restrict__ W_namp, const float* __restrict__ b_namp,
                             const float* __restrict__ W_B,    const float* __restrict__ b_B,
                             float* __restrict__ Wc, float* __restrict__ bc) {
  int n = blockIdx.x * blockDim.x + threadIdx.x;
  if (n >= NHEADP) return;
  for (int d = 0; d < 128; ++d) {
    float w = 0.0f;
    if (n < 256)            { int c = n / 128, j = n % 128; w = W_harm[((size_t)c * 128 + d) * 128 + j]; }
    else if (n < 258)       { int c = n - 256;              w = W_amp[c * 128 + d]; }
    else if (n < 258 + 1026){ int m = n - 258; int c = m / 513, j = m % 513;
                              w = W_noise[((size_t)c * 128 + d) * 513 + j]; }
    else if (n < 1286)      { int c = n - 1284;             w = W_namp[c * 128 + d]; }
    else if (n == 1286)     {                               w = W_B[d]; }
    Wc[(size_t)d * NHEADP + n] = w;
  }
  float bb = 0.0f;
  if (n < 256)       bb = b_harm[n];
  else if (n < 258)  bb = b_amp[n - 256];
  else if (n < 1284) bb = b_noise[n - 258];
  else if (n < 1286) bb = b_namp[n - 1284];
  else if (n == 1286) bb = b_B[0];
  bc[n] = bb;
}

// ------------------------------------------- head nonlinearities per (b,t)
__global__ void k_heads(const float* __restrict__ raw, float* __restrict__ harm_amps,
                        float* __restrict__ nmag, float* __restrict__ sB) {
  int bt = blockIdx.x;
  const float* r = raw + (size_t)bt * NHEADP;
  int tid = threadIdx.x;                         // 128
  __shared__ float red[128];
  for (int c = 0; c < 2; ++c) {
    float x = r[c * 128 + tid];
    red[tid] = x; __syncthreads();
    for (int s = 64; s > 0; s >>= 1) { if (tid < s) red[tid] = fmaxf(red[tid], red[tid + s]); __syncthreads(); }
    float mx = red[0]; __syncthreads();
    float e = __expf(x - mx);
    red[tid] = e; __syncthreads();
    for (int s = 64; s > 0; s >>= 1) { if (tid < s) red[tid] += red[tid + s]; __syncthreads(); }
    float sum = red[0]; __syncthreads();
    float al = r[256 + c];
    float amp = (al > 20.0f) ? al : log1pf(__expf(al));   // softplus
    harm_amps[(((size_t)c * BATCH * TFRM) + bt) * 128 + tid] = (e / sum) * amp;
    float nl = r[1284 + c];
    float namp = (nl > 20.0f) ? nl : log1pf(__expf(nl));
    for (int j = tid; j < 513; j += 128) {
      float sg = 1.0f / (1.0f + __expf(-r[258 + c * 513 + j]));
      nmag[(((size_t)c * BATCH * TFRM) + bt) * 513 + j] = sg * namp;
    }
    __syncthreads();
  }
  if (tid == 0) sB[bt] = 1.0f / (1.0f + __expf(-r[1286]));
}

// -------------------- phase accumulator: f0 upsample + cumsum scan per batch
__global__ void k_phase(const float* __restrict__ f0, float* __restrict__ f0_up,
                        float* __restrict__ phase, float* __restrict__ f0umean,
                        float* __restrict__ n_active) {
  int b = blockIdx.x, tid = threadIdx.x;         // 1024 threads
  __shared__ float sc[1024];
  __shared__ float carry;
  if (tid == 0) carry = 0.0f;
  __syncthreads();
  const float scale = (float)TFRM / (float)NSAMP;
  for (int base = 0; base < NSAMP; base += 1024) {
    int s = base + tid;
    float v = 0.0f;
    if (s < NSAMP) {
      float pos = ((float)s + 0.5f) * scale - 0.5f;
      pos = fminf(fmaxf(pos, 0.0f), (float)(TFRM - 1));
      int lo = (int)floorf(pos); int hi = min(lo + 1, TFRM - 1);
      float fr = pos - (float)lo;
      float fu = f0[b * TFRM + lo] * (1.0f - fr) + f0[b * TFRM + hi] * fr;
      f0_up[(size_t)b * NSAMP + s] = fu;
      v = 2.0f * PI_F * fu / SR_F;
    }
    sc[tid] = v; __syncthreads();
    for (int off = 1; off < 1024; off <<= 1) {
      float add = (tid >= off) ? sc[tid - off] : 0.0f;
      __syncthreads();
      sc[tid] += add;
      __syncthreads();
    }
    float cbase = carry;
    float inc = sc[tid];
    __syncthreads();
    if (s < NSAMP) phase[(size_t)b * NSAMP + s] = cbase + inc;
    if (tid == 1023) carry = cbase + inc;
    __syncthreads();
  }
  if (tid == 0) {
    float fmean = carry * SR_F / (2.0f * PI_F) / (float)NSAMP;
    f0umean[b] = fmean;
    int cnt = 0;
    for (int k = 1; k <= 128; ++k) if ((float)k * fmean < SR_F * 0.45f) cnt++;
    n_active[b] = (float)max(cnt, 1);
  }
}

// ------------------- per-batch scalars: inharmonicity + stretch + f0 clip mean
__global__ void k_scalars(const float* __restrict__ sB, const float* __restrict__ f0,
                          float* __restrict__ inh, float* __restrict__ stretch,
                          float* __restrict__ f0_cmean) {
  int b = blockIdx.x, tid = threadIdx.x;         // 256
  __shared__ float r1[256], r2[256];
  __shared__ float inh_s;
  float s1 = 0.0f, s2 = 0.0f;
  for (int t = tid; t < TFRM; t += 256) {
    s1 += sB[b * TFRM + t];
    s2 += fmaxf(f0[b * TFRM + t], 20.0f);
  }
  r1[tid] = s1; r2[tid] = s2; __syncthreads();
  for (int s = 128; s > 0; s >>= 1) {
    if (tid < s) { r1[tid] += r1[tid + s]; r2[tid] += r2[tid + s]; }
    __syncthreads();
  }
  if (tid == 0) {
    float msB = r1[0] / (float)TFRM;
    float f0m = r2[0] / (float)TFRM;
    f0_cmean[b] = f0m;
    float midi = 69.0f + 12.0f * log2f(f0m / 440.0f);
    float bmax = 0.0008f * __expf(-(midi - 21.0f) / 88.0f * __logf(10.0f));
    inh_s = msB * bmax;
    inh[b] = inh_s;
  }
  __syncthreads();
  if (tid < 128) {
    float k = (float)(tid + 1);
    stretch[b * 128 + tid] = sqrtf(1.0f + inh_s * k * k);
  }
}

// ---------------------------------------- harmonic additive synth (L and R)
__global__ void k_harm(const float* __restrict__ harm_amps, const float* __restrict__ f0_up,
                       const float* __restrict__ phase, const float* __restrict__ stretch,
                       const float* __restrict__ n_active,
                       float* __restrict__ harmL, float* __restrict__ harmR) {
  int idx = blockIdx.x * blockDim.x + threadIdx.x;    // over B*S
  if (idx >= BATCH * NSAMP) return;
  int b = idx / NSAMP, s = idx % NSAMP;
  float fu = f0_up[idx];
  float ph = phase[idx];
  float pos = ((float)s + 0.5f) * (float)TFRM / (float)NSAMP - 0.5f;
  pos = fminf(fmaxf(pos, 0.0f), (float)(TFRM - 1));
  int lo = (int)floorf(pos); int hi = min(lo + 1, TFRM - 1);
  float fr = pos - (float)lo;
  const float* aL0 = harm_amps + ((size_t)(0 * BATCH + b) * TFRM + lo) * 128;
  const float* aL1 = harm_amps + ((size_t)(0 * BATCH + b) * TFRM + hi) * 128;
  const float* aR0 = harm_amps + ((size_t)(1 * BATCH + b) * TFRM + lo) * 128;
  const float* aR1 = harm_amps + ((size_t)(1 * BATCH + b) * TFRM + hi) * 128;
  const float* st = stretch + b * 128;
  float accL = 0.0f, accR = 0.0f;
  const float lim = SR_F * 0.45f;
  const float inv2pi = 1.0f / (2.0f * PI_F);
  #pragma unroll 4
  for (int k = 0; k < 128; ++k) {
    float fk = (float)(k + 1) * st[k];
    if (fu * fk < lim) {
      float arg = fk * ph;
      arg -= truncf(arg * inv2pi) * (2.0f * PI_F);    // range-reduce
      float sn = __sinf(arg);                          // shared by L and R
      accL += (aL0[k] + fr * (aL1[k] - aL0[k])) * sn;
      accR += (aR0[k] + fr * (aR1[k] - aR0[k])) * sn;
    }
  }
  float na = n_active[b];
  harmL[idx] = accL / na;
  harmR[idx] = accR / na;
}

// ----------------------------------------------------- 1024-pt FFT in LDS
__device__ __forceinline__ unsigned bitrev10(unsigned x) { return __brev(x) >> 22; }

__device__ void fft1024(float* re, float* im, int inverse) {
  int tid = threadIdx.x;                    // 256
  for (int len = 2; len <= 1024; len <<= 1) {
    int half = len >> 1;
    for (int i = tid; i < 512; i += 256) {
      int grp = i / half, pos = i % half;
      int a = grp * len + pos, b2 = a + half;
      float ang = (inverse ? 2.0f : -2.0f) * PI_F * (float)pos / (float)len;
      float wr = __cosf(ang), wi = __sinf(ang);
      float br = re[b2], bi = im[b2];
      float tr = wr * br - wi * bi;
      float ti = wr * bi + wi * br;
      float ar = re[a], ai = im[a];
      re[a] = ar + tr;  im[a] = ai + ti;
      re[b2] = ar - tr; im[b2] = ai - ti;
    }
    __syncthreads();
  }
}

// ------------------- per-frame: STFT -> spectral shaping -> iSTFT -> windowed
__global__ void k_noise_frame(const float* __restrict__ noiseL, const float* __restrict__ noiseR,
                              const float* __restrict__ nmag, const float* __restrict__ f0_cmean,
                              float* __restrict__ tframes) {
  int bid = blockIdx.x;                     // (c*B + b)*TSTFT + tf
  int tf = bid % TSTFT; int cb = bid / TSTFT; int b = cb % BATCH; int c = cb / BATCH;
  __shared__ float re[NFFT], im[NFFT], yr[NFFT], yi[NFFT];
  __shared__ float mag[513];
  int tid = threadIdx.x;                    // 256
  const float* noise = (c == 0) ? noiseL : noiseR;

  for (int j = tid; j < NFFT; j += 256) {
    int n = tf * HOPN + j - PADN;           // index into original signal
    int m = n;
    if (m < 0) m = -m;                      // reflect pad
    if (m >= NSAMP) m = 2 * NSAMP - 2 - m;
    float w = 0.5f - 0.5f * __cosf(2.0f * PI_F * (float)j / (float)NFFT);
    unsigned rb = bitrev10((unsigned)j);
    re[rb] = noise[(size_t)b * NSAMP + m] * w;
    im[rb] = 0.0f;
  }
  float posT = ((float)tf + 0.5f) * (float)TFRM / (float)TSTFT - 0.5f;
  posT = fminf(fmaxf(posT, 0.0f), (float)(TFRM - 1));
  int tl = (int)floorf(posT); int th = min(tl + 1, TFRM - 1);
  float ft = posT - (float)tl;
  float f0m = f0_cmean[b];
  const float* nm_l = nmag + ((size_t)(c * BATCH + b) * TFRM + tl) * 513;
  const float* nm_h = nmag + ((size_t)(c * BATCH + b) * TFRM + th) * 513;
  for (int j = tid; j < 513; j += 256) {
    float freq = (float)j * (SR_F / (float)NFFT);
    float src = freq / f0m * (513.0f / 128.0f);
    src = fminf(fmaxf(src, 0.0f), 512.0f);
    int lo = (int)src; int hi = min(lo + 1, 512);
    float fr = src - (float)lo;
    float m_lo = nm_l[lo] + ft * (nm_h[lo] - nm_l[lo]);
    float m_hi = nm_l[hi] + ft * (nm_h[hi] - nm_l[hi]);
    mag[j] = m_lo + fr * (m_hi - m_lo);
  }
  __syncthreads();
  fft1024(re, im, 0);                       // natural-order spectrum
  for (int k = tid; k <= 512; k += 256) {
    float mgv = mag[k];
    float Yr = re[k] * mgv, Yi = im[k] * mgv;
    yr[bitrev10((unsigned)k)] = Yr;
    yi[bitrev10((unsigned)k)] = Yi;
    if (k >= 1 && k <= 511) {
      unsigned k2 = 1024u - (unsigned)k;
      yr[bitrev10(k2)] = Yr;
      yi[bitrev10(k2)] = -Yi;
    }
  }
  __syncthreads();
  fft1024(yr, yi, 1);                       // inverse (unscaled)
  for (int j = tid; j < NFFT; j += 256) {
    float w = 0.5f - 0.5f * __cosf(2.0f * PI_F * (float)j / (float)NFFT);
    tframes[(size_t)bid * NFFT + j] = yr[j] * (1.0f / (float)NFFT) * w;
  }
}

// --------------------- gather-OLA (deterministic) + loudness + final output
__global__ void k_combine(const float* __restrict__ tframes, const float* __restrict__ harmL,
                          const float* __restrict__ harmR, const float* __restrict__ loud,
                          float* __restrict__ out) {
  int idx = blockIdx.x * blockDim.x + threadIdx.x;    // over B*S
  if (idx >= BATCH * NSAMP) return;
  int b = idx / NSAMP, s = idx % NSAMP;
  int n = s + PADN;
  int t0 = (n - (NFFT - 1) + (HOPN - 1)) / HOPN; if (t0 < 0) t0 = 0;
  int t1 = n / HOPN; if (t1 > TSTFT - 1) t1 = TSTFT - 1;
  float wsq = 0.0f, accL = 0.0f, accR = 0.0f;
  for (int t = t0; t <= t1; ++t) {
    int j = n - t * HOPN;
    float w = 0.5f - 0.5f * __cosf(2.0f * PI_F * (float)j / (float)NFFT);
    wsq += w * w;
    accL += tframes[((size_t)(0 * BATCH + b) * TSTFT + t) * NFFT + j];
    accR += tframes[((size_t)(1 * BATCH + b) * TSTFT + t) * NFFT + j];
  }
  float inv = (wsq > 1e-11f) ? (1.0f / wsq) : 1.0f;
  float pos = ((float)s + 0.5f) * (float)TFRM / (float)NSAMP - 0.5f;
  pos = fminf(fmaxf(pos, 0.0f), (float)(TFRM - 1));
  int lo = (int)floorf(pos); int hi = min(lo + 1, TFRM - 1);
  float fr = pos - (float)lo;
  float l0 = __powf(10.0f, loud[b * TFRM + lo] * (1.0f / 20.0f));
  float l1 = __powf(10.0f, loud[b * TFRM + hi] * (1.0f / 20.0f));
  float lu = l0 + fr * (l1 - l0);
  out[((size_t)b * 2 + 0) * NSAMP + s] = (harmL[idx] + accL * inv) * lu;
  out[((size_t)b * 2 + 1) * NSAMP + s] = (harmR[idx] + accR * inv) * lu;
}

// ===========================================================================
extern "C" void kernel_launch(void* const* d_in, const int* in_sizes, int n_in,
                              void* d_out, int out_size, void* d_ws, size_t ws_size,
                              hipStream_t stream) {
  (void)in_sizes; (void)n_in; (void)out_size; (void)ws_size;
  const float* f0      = (const float*)d_in[0];
  const float* loud    = (const float*)d_in[1];
  const float* noiseL  = (const float*)d_in[2];
  const float* noiseR  = (const float*)d_in[3];
  const float* W_pre1  = (const float*)d_in[4];
  const float* b_pre1  = (const float*)d_in[5];
  const float* W_pre2  = (const float*)d_in[6];
  const float* b_pre2  = (const float*)d_in[7];
  const float* W_ih    = (const float*)d_in[8];
  const float* W_hh    = (const float*)d_in[9];
  const float* b_ih    = (const float*)d_in[10];
  const float* b_hh    = (const float*)d_in[11];
  const float* W_post  = (const float*)d_in[12];
  const float* b_post  = (const float*)d_in[13];
  const float* W_harm  = (const float*)d_in[14];
  const float* b_harm  = (const float*)d_in[15];
  const float* W_amp   = (const float*)d_in[16];
  const float* b_amp   = (const float*)d_in[17];
  const float* W_noise = (const float*)d_in[18];
  const float* b_noise = (const float*)d_in[19];
  const float* W_namp  = (const float*)d_in[20];
  const float* b_namp  = (const float*)d_in[21];
  const float* W_B     = (const float*)d_in[22];
  const float* b_B     = (const float*)d_in[23];
  const int*   vel     = (const int*)  d_in[24];

  const int M = BATCH * TFRM;     // 800
  float* ws = (float*)d_ws;
  size_t off = 0;
  auto alloc = [&](size_t n) { float* p = ws + off; off += n; return p; };

  float* feat0     = alloc((size_t)M * KFEAT);
  float* W1p       = alloc((size_t)KFEAT * 128);
  float* h1        = alloc((size_t)M * 128);
  float* h2        = alloc((size_t)M * 128);
  float* gi        = alloc((size_t)M * 192);
  float* hs        = alloc((size_t)M * 64);
  float* postf     = alloc((size_t)M * 128);
  float* Wc        = alloc((size_t)128 * NHEADP);
  float* bc        = alloc(NHEADP);
  float* headraw   = alloc((size_t)M * NHEADP);
  float* harm_amps = alloc((size_t)2 * M * 128);
  float* nmag      = alloc((size_t)2 * M * 513);
  float* sB        = alloc(M);
  float* f0_up     = alloc((size_t)BATCH * NSAMP);
  float* phase     = alloc((size_t)BATCH * NSAMP);
  float* f0umean   = alloc(BATCH);
  float* n_active  = alloc(BATCH);
  float* inh       = alloc(BATCH);
  float* f0_cmean  = alloc(BATCH);
  float* stretch   = alloc(BATCH * 128);
  float* harmL     = alloc((size_t)BATCH * NSAMP);
  float* harmR     = alloc((size_t)BATCH * NSAMP);
  float* tframes   = alloc((size_t)2 * BATCH * TSTFT * NFFT);

  auto strips = [](int m, int n) { return (m / 16) * (n / 32); };

  k_encode<<<(M + 255) / 256, 256, 0, stream>>>(f0, vel, feat0);
  k_pad_w1<<<(KFEAT * 128 + 255) / 256, 256, 0, stream>>>(W_pre1, W1p);

  k_gemm_wmma<<<strips(M, 128), 32, 0, stream>>>(feat0, W1p, b_pre1, h1, M, KFEAT, 128, 0, 1);
  k_gemm_wmma<<<strips(M, 128), 32, 0, stream>>>(h1, W_pre2, b_pre2, h2, M, 128, 128, 0, 1);
  k_gemm_wmma<<<strips(M, 192), 32, 0, stream>>>(h2, W_ih, b_ih, gi, M, 128, 192, 1, 0);

  k_gru<<<1, 384, 0, stream>>>(gi, W_hh, b_hh, hs);

  k_gemm_wmma<<<strips(M, 128), 32, 0, stream>>>(hs, W_post, b_post, postf, M, 64, 128, 0, 1);

  k_pack_heads<<<(NHEADP + 255) / 256, 256, 0, stream>>>(W_harm, b_harm, W_amp, b_amp,
                                                         W_noise, b_noise, W_namp, b_namp,
                                                         W_B, b_B, Wc, bc);
  k_gemm_wmma<<<strips(M, NHEADP), 32, 0, stream>>>(postf, Wc, bc, headraw, M, 128, NHEADP, 0, 0);

  k_heads<<<M, 128, 0, stream>>>(headraw, harm_amps, nmag, sB);

  k_phase<<<BATCH, 1024, 0, stream>>>(f0, f0_up, phase, f0umean, n_active);
  k_scalars<<<BATCH, 256, 0, stream>>>(sB, f0, inh, stretch, f0_cmean);

  k_harm<<<(BATCH * NSAMP + 255) / 256, 256, 0, stream>>>(harm_amps, f0_up, phase, stretch,
                                                          n_active, harmL, harmR);

  k_noise_frame<<<2 * BATCH * TSTFT, 256, 0, stream>>>(noiseL, noiseR, nmag, f0_cmean, tframes);

  k_combine<<<(BATCH * NSAMP + 255) / 256, 256, 0, stream>>>(tframes, harmL, harmR, loud,
                                                             (float*)d_out);
}
